// SSAccuracyLoss2_46325517254719
// MI455X (gfx1250) — compile-verified, matrix operand order checked by reading, not run on hardware
//
#include <hip/hip_runtime.h>

typedef __attribute__((ext_vector_type(4)))  float    v4f;
typedef __attribute__((ext_vector_type(16))) _Float16 v16h;
typedef __attribute__((ext_vector_type(8)))  float    v8f;

// Each wave handles 512 consecutive rows: lane L takes rows base + L + 32*k, k=0..15.
// Row = 8 floats of real + 8 floats of pred (32B each) -> two B128 loads per array.

__global__ void ssacc_init_ws(unsigned int* ws) {
    if (threadIdx.x == 0 && blockIdx.x == 0) ws[0] = 0u;
}

__global__ __launch_bounds__(256) void ssacc_count_kernel(
        const float* __restrict__ real,
        const float* __restrict__ pred,
        unsigned int* __restrict__ ws) {
    const int lane      = threadIdx.x & 31;
    const int waveInBlk = threadIdx.x >> 5;
    const long long waveId = (long long)blockIdx.x * (blockDim.x >> 5) + waveInBlk;
    const long long base   = waveId * 512;  // 32 lanes * 16 rows

    v16h a;  // 16 row-correct flags per lane, as f16 0.0/1.0
#pragma unroll
    for (int k = 0; k < 16; ++k) {
        const long long r  = base + lane + 32 * k;
        const v4f* rp = (const v4f*)(real + r * 8);
        const v4f* pp = (const v4f*)(pred + r * 8);
        v4f r0 = __builtin_nontemporal_load(rp);
        v4f r1 = __builtin_nontemporal_load(rp + 1);
        v4f p0 = __builtin_nontemporal_load(pp);
        v4f p1 = __builtin_nontemporal_load(pp + 1);
        // real is exactly 0.0 or 1.0; match iff (pred>0.5) == (real!=0)
        bool ok = ((p0.x > 0.5f) == (r0.x != 0.0f))
                & ((p0.y > 0.5f) == (r0.y != 0.0f))
                & ((p0.z > 0.5f) == (r0.z != 0.0f))
                & ((p0.w > 0.5f) == (r0.w != 0.0f))
                & ((p1.x > 0.5f) == (r1.x != 0.0f))
                & ((p1.y > 0.5f) == (r1.y != 0.0f))
                & ((p1.z > 0.5f) == (r1.z != 0.0f))
                & ((p1.w > 0.5f) == (r1.w != 0.0f));
        a[k] = ok ? (_Float16)1.0f : (_Float16)0.0f;
    }

    // All-ones B: D[m][n] = rowSum_A(m) for every n, so sum(D) = 16 * sum(flags).
    v16h b;
#pragma unroll
    for (int k = 0; k < 16; ++k) b[k] = (_Float16)1.0f;

    v8f c = {};
    c = __builtin_amdgcn_wmma_f32_16x16x32_f16(
            /*neg_a=*/false, a, /*neg_b=*/false, b,
            /*c_mod=*/(short)0, c, /*reuse_a=*/false, /*reuse_b=*/false);

    float s = c[0] + c[1] + c[2] + c[3] + c[4] + c[5] + c[6] + c[7];

    // wave32 reduce
#pragma unroll
    for (int off = 16; off > 0; off >>= 1)
        s += __shfl_down(s, off, 32);

    __shared__ float wsum[8];
    if (lane == 0) wsum[waveInBlk] = s;
    __syncthreads();
    if (threadIdx.x == 0) {
        float t = 0.0f;
        const int nw = (int)(blockDim.x >> 5);
        for (int i = 0; i < nw; ++i) t += wsum[i];
        // t is an exact integer (<= 16*4096 per block)
        atomicAdd(ws, (unsigned int)(t + 0.5f));
    }
}

__global__ void ssacc_finalize(const unsigned int* __restrict__ ws,
                               float* __restrict__ out, float inv) {
    if (threadIdx.x == 0 && blockIdx.x == 0)
        out[0] = 1.0f - (float)ws[0] * inv;
}

extern "C" void kernel_launch(void* const* d_in, const int* in_sizes, int n_in,
                              void* d_out, int out_size, void* d_ws, size_t ws_size,
                              hipStream_t stream) {
    const float* real = (const float*)d_in[0];
    const float* pred = (const float*)d_in[1];
    float* out = (float*)d_out;
    unsigned int* cnt = (unsigned int*)d_ws;

    const long long n    = (long long)in_sizes[0];  // B*L*C
    const long long rows = n / 8;                   // C == 8
    // 512 rows per wave, 8 waves per block (exact fit for 1,048,576 rows -> 256 blocks)
    int blocks = (int)(rows / (512 * 8));
    if (blocks < 1) blocks = 1;

    ssacc_init_ws<<<1, 1, 0, stream>>>(cnt);
    ssacc_count_kernel<<<blocks, 256, 0, stream>>>(real, pred, cnt);

    const float inv = 1.0f / (16.0f * (float)rows);  // WMMA replicates row sums 16x
    ssacc_finalize<<<1, 1, 0, stream>>>(cnt, out, inv);
}
